// AudioCodec_87282325389770
// MI455X (gfx1250) — compile-verified, matrix operand order checked by reading, not run on hardware
//
#include <hip/hip_runtime.h>
#include <hip/hip_bf16.h>
#include <float.h>

typedef __attribute__((ext_vector_type(16))) __bf16 v16bf;
typedef __attribute__((ext_vector_type(8)))  float  v8f;
typedef int v4i __attribute__((vector_size(4 * sizeof(int))));
typedef __attribute__((address_space(1))) v4i* gptr4;   // global (device) int4*
typedef __attribute__((address_space(3))) v4i* lptr4;   // LDS int4*

union FragH { v16bf v; unsigned int u[8]; };

__device__ __forceinline__ unsigned short f2bf(float f) {
  unsigned int u = __float_as_uint(f);
  u += 0x7FFFu + ((u >> 16) & 1u);          // round-to-nearest-even
  return (unsigned short)(u >> 16);
}

// 16-bit A/B fragment K-mapping per CDNA5 ISA 7.12.2 (16x32 A-matrix):
// lane<16: VGPR0-3 hold K=0..7 (pairs), VGPR4-7 hold K=16..23
// lane>=16: VGPR0-3 hold K=8..15, VGPR4-7 hold K=24..31
__device__ __forceinline__ int kmap(int j, int h) {
  return (j < 4) ? (2 * j + 8 * h) : (16 + 2 * (j - 4) + 8 * h);
}

// ---------------------------------------------------------------------------
// Implicit-GEMM conv1d / transposed-conv1d with WMMA bf16->f32.
// Block: 256 threads (8 waves). Macro-tile: 64 (Cout) x 64 (Tout); each wave
// computes a 16x32 strip (two 16x16 WMMA tiles, two accumulators).
// K ordering: kdim = kw*Cin + ci (Cin is a power of two for every layer).
// Weight tiles staged with async global->LDS B128 copies when fully dense.
// Transposed conv: ostride=1, lhsDil=stride, padEff=K-1-pad.
// act: 0=none, 1=elu, 2=tanh. xTmajor: x laid out [t][ci] instead of [ci][t].
// ---------------------------------------------------------------------------
__global__ __launch_bounds__(256) void k_conv_wmma(
    const float* __restrict__ x, const unsigned short* __restrict__ A,
    const float* __restrict__ bias, float* __restrict__ y,
    int Cin, int Tin, int Cout, int Tout, int K,
    int ostride, int lhsDil, int padEff, int act, int xTmajor)
{
  __shared__ unsigned short As[64 * 32];   // row stride 32 (async-friendly)
  __shared__ unsigned short Bs[64 * 34];   // row stride 34 (bank-conflict pad)
  const int tid = threadIdx.x;
  const int t0 = blockIdx.x * 64;
  const int m0 = blockIdx.y * 64;
  const int bz = blockIdx.z;
  const int Kdim = Cin * K;
  const int cshift = 31 - __clz(Cin);      // log2(Cin)
  const int cmask = Cin - 1;
  const long xbase = (long)bz * (long)Cin * (long)Tin;
  const bool asyncA = ((Kdim & 31) == 0) && (Cout >= 64);

  const int wave = tid >> 5, lane = tid & 31;
  const int h = lane >> 4, r = lane & 15;
  const int wm = (wave & 3) * 16;    // 4 tiles along Cout
  const int wn = (wave >> 2) * 32;   // 2x(16x16) strip along Tout
  v8f acc0 = {0.f, 0.f, 0.f, 0.f, 0.f, 0.f, 0.f, 0.f};
  v8f acc1 = {0.f, 0.f, 0.f, 0.f, 0.f, 0.f, 0.f, 0.f};

  // async A staging: thread -> (row am, 16-byte chunk ac)
  const int am = tid >> 2, ac = (tid & 3) * 8;

  for (int k0 = 0; k0 < Kdim; k0 += 32) {
    if (asyncA) {
      // Dense, aligned, in-bounds: one async B128 global->LDS copy per thread
      const unsigned short* gp = A + (size_t)(m0 + am) * Kdim + k0 + ac;
      unsigned short* lp = &As[am * 32 + ac];
      __builtin_amdgcn_global_load_async_to_lds_b128((gptr4)gp, (lptr4)lp, 0, 0);
    } else {
      for (int e = tid; e < 64 * 32; e += 256) {
        int m = e >> 5, kk = e & 31, kg = k0 + kk;
        unsigned short v = 0;
        if (m0 + m < Cout && kg < Kdim) v = A[(size_t)(m0 + m) * Kdim + kg];
        As[e] = v;
      }
    }
    // B staging, kk-major so consecutive lanes sweep consecutive t (coalesced)
    for (int e = tid; e < 64 * 32; e += 256) {
      int kk = e >> 6, n = e & 63, kg = k0 + kk;
      float val = 0.f;
      if (kg < Kdim) {
        int kw = kg >> cshift, ci = kg & cmask;
        int pos = (t0 + n) * ostride + kw - padEff;
        int src = pos;
        bool ok;
        if (lhsDil == 1) {
          ok = (pos >= 0) && (pos < Tin);
        } else {
          ok = (pos >= 0) && ((pos % lhsDil) == 0);
          src = pos / lhsDil;
          ok = ok && (src < Tin);
        }
        if (ok)
          val = xTmajor ? x[xbase + (long)src * Cin + ci]
                        : x[xbase + (long)ci * Tin + src];
      }
      Bs[n * 34 + kk] = f2bf(val);
    }
    if (asyncA) asm volatile("s_wait_asynccnt 0x0" ::: "memory");
    __syncthreads();

    FragH a, b0, b1;
#pragma unroll
    for (int j = 0; j < 8; ++j) {
      int kb = kmap(j, h);
      a.u[j]  = *(const unsigned int*)&As[(wm + r) * 32 + kb];
      b0.u[j] = *(const unsigned int*)&Bs[(wn + r) * 34 + kb];
      b1.u[j] = *(const unsigned int*)&Bs[(wn + 16 + r) * 34 + kb];
    }
    acc0 = __builtin_amdgcn_wmma_f32_16x16x32_bf16(false, a.v, false, b0.v,
                                                   (short)0, acc0, false, false);
    acc1 = __builtin_amdgcn_wmma_f32_16x16x32_bf16(false, a.v, false, b1.v,
                                                   (short)0, acc1, false, false);
    __syncthreads();
  }

  const long ybase = (long)bz * (long)Cout * (long)Tout;
  const int ta = t0 + wn + r;
  const int tb = ta + 16;
#pragma unroll
  for (int rr = 0; rr < 8; ++rr) {
    int co = m0 + wm + rr + 8 * h;
    if (co < Cout) {
      float bi = bias[co];
      float v0 = acc0[rr] + bi;
      float v1 = acc1[rr] + bi;
      if (act == 1) {
        v0 = v0 > 0.f ? v0 : (__expf(v0) - 1.f);
        v1 = v1 > 0.f ? v1 : (__expf(v1) - 1.f);
      } else if (act == 2) {
        v0 = tanhf(v0);
        v1 = tanhf(v1);
      }
      if (ta < Tout) y[ybase + (long)co * Tout + ta] = v0;
      if (tb < Tout) y[ybase + (long)co * Tout + tb] = v1;
    }
  }
}

// ---------------------------------------------------------------------------
// VQ: per 16 time-rows, argmin over 1024 codes of (|c|^2 - 2 x.c) via WMMA.
// ---------------------------------------------------------------------------
__global__ __launch_bounds__(256) void k_vq(
    const float* __restrict__ a5,            // (8,512,8192) channel-major
    const unsigned short* __restrict__ cbf,  // 1024x512 bf16
    const float* __restrict__ cb,            // 1024x512 f32
    const float* __restrict__ cnorm,         // 1024
    float* __restrict__ qout,                // (65536,512)
    float* __restrict__ codes,               // 65536 (as float)
    float* __restrict__ partial)             // per-block commitment sums
{
  __shared__ unsigned short embS[16 * 512];
  __shared__ float minS[8 * 16];
  __shared__ int   idxS[8 * 16];
  __shared__ int   idx16[16];
  __shared__ float red[256];

  const int tid = threadIdx.x;
  const int gt0 = blockIdx.x * 16;
  const int b = gt0 >> 13;       // 8192 t per batch
  const int tq0 = gt0 & 8191;
  const long abase = (long)b * 512 * 8192;

  for (int e = tid; e < 16 * 512; e += 256) {
    int tl = e >> 9, d = e & 511;
    embS[tl * 512 + d] = f2bf(a5[abase + (long)d * 8192 + tq0 + tl]);
  }
  __syncthreads();

  const int wave = tid >> 5, lane = tid & 31;
  const int h = lane >> 4, r = lane & 15;
  float minv = FLT_MAX;
  int mini = 0x7FFFFFFF;

  for (int ct = wave; ct < 64; ct += 8) {   // 8 code-tiles per wave, uniform
    int c0 = ct * 16;
    v8f acc = {0.f, 0.f, 0.f, 0.f, 0.f, 0.f, 0.f, 0.f};
    for (int ks = 0; ks < 16; ++ks) {       // D=512 in 16 k-steps
      FragH a, bf;
#pragma unroll
      for (int j = 0; j < 8; ++j) {
        int d = ks * 32 + kmap(j, h);
        a.u[j]  = *(const unsigned int*)&cbf[(c0 + r) * 512 + d];
        bf.u[j] = *(const unsigned int*)&embS[r * 512 + d];
      }
      acc = __builtin_amdgcn_wmma_f32_16x16x32_bf16(false, a.v, false, bf.v,
                                                    (short)0, acc, false, false);
    }
#pragma unroll
    for (int rr = 0; rr < 8; ++rr) {
      int c = c0 + rr + 8 * h;
      float s = cnorm[c] - 2.f * acc[rr];
      if (s < minv || (s == minv && c < mini)) { minv = s; mini = c; }
    }
  }
  // fold lane L with L+16 (same t, different code subsets)
  {
    float ov = __shfl_xor(minv, 16, 32);
    int   oi = __shfl_xor(mini, 16, 32);
    if (ov < minv || (ov == minv && oi < mini)) { minv = ov; mini = oi; }
  }
  if (lane < 16) { minS[wave * 16 + r] = minv; idxS[wave * 16 + r] = mini; }
  __syncthreads();
  if (tid < 16) {
    float bv = FLT_MAX; int bi = 0x7FFFFFFF;
    for (int w = 0; w < 8; ++w) {
      float v = minS[w * 16 + tid]; int i = idxS[w * 16 + tid];
      if (v < bv || (v == bv && i < bi)) { bv = v; bi = i; }
    }
    idx16[tid] = bi;
    codes[gt0 + tid] = (float)bi;
  }
  __syncthreads();
  // gather f32 codebook rows -> quantized output; accumulate commitment loss
  float lsum = 0.f;
  for (int e = tid; e < 16 * 512; e += 256) {
    int tl = e >> 9, d = e & 511;
    int ix = idx16[tl];
    float qv = cb[ix * 512 + d];
    qout[(long)(gt0 + tl) * 512 + d] = qv;
    float ev = a5[abase + (long)d * 8192 + tq0 + tl];
    float df = qv - ev;
    lsum += df * df;
  }
  red[tid] = lsum;
  __syncthreads();
  for (int s = 128; s > 0; s >>= 1) {
    if (tid < s) red[tid] += red[tid + s];
    __syncthreads();
  }
  if (tid == 0) partial[blockIdx.x] = red[0];
}

// --------------------------- prep / reduction kernels ----------------------
__global__ void k_cvt_flat(const float* __restrict__ w,
                           unsigned short* __restrict__ o, int n) {
  int i = blockIdx.x * 256 + threadIdx.x;
  if (i < n) o[i] = f2bf(w[i]);
}

// encoder weights (Cout,Cin,K) -> A[co][kw*Cin+ci]
__global__ void k_cvt_enc(const float* __restrict__ w,
                          unsigned short* __restrict__ o,
                          int Cout, int Cin, int K) {
  int Kd = Cin * K, n = Cout * Kd;
  for (int e = blockIdx.x * 256 + threadIdx.x; e < n; e += gridDim.x * 256) {
    int co = e / Kd, rem = e - co * Kd;
    int kw = rem / Cin, ci = rem - kw * Cin;
    o[e] = f2bf(w[(co * Cin + ci) * K + kw]);
  }
}

// decoder weights stored (Cin, Cout, K); effective conv weight is
// A[co][kw*Cin+ci] = w[ci][co][K-1-kw]
__global__ void k_cvt_dec(const float* __restrict__ w,
                          unsigned short* __restrict__ o,
                          int Cout, int Cin, int K) {
  int Kd = Cin * K, n = Cout * Kd;
  for (int e = blockIdx.x * 256 + threadIdx.x; e < n; e += gridDim.x * 256) {
    int co = e / Kd, rem = e - co * Kd;
    int kw = rem / Cin, ci = rem - kw * Cin;
    o[e] = f2bf(w[(ci * Cout + co) * K + (K - 1 - kw)]);
  }
}

__global__ void k_cb_norm(const float* __restrict__ cb, float* __restrict__ cn) {
  int c = blockIdx.x * 256 + threadIdx.x;
  if (c < 1024) {
    float s = 0.f;
    for (int d = 0; d < 512; ++d) { float v = cb[c * 512 + d]; s += v * v; }
    cn[c] = s;
  }
}

__global__ __launch_bounds__(256) void k_recon_loss(
    const float* __restrict__ rec, const float* __restrict__ wav,
    float* __restrict__ partial, int n) {
  __shared__ float red[256];
  float s = 0.f;
  for (int i = blockIdx.x * 256 + threadIdx.x; i < n; i += gridDim.x * 256)
    s += fabsf(rec[i] - wav[i]);
  red[threadIdx.x] = s;
  __syncthreads();
  for (int st = 128; st > 0; st >>= 1) {
    if (threadIdx.x < st) red[threadIdx.x] += red[threadIdx.x + st];
    __syncthreads();
  }
  if (threadIdx.x == 0) partial[blockIdx.x] = red[0];
}

__global__ __launch_bounds__(256) void k_finalize(
    const float* __restrict__ p1, int n1, const float* __restrict__ p2, int n2,
    float* __restrict__ out_commit, float* __restrict__ out_recon) {
  __shared__ float r1[256], r2[256];
  float a = 0.f, b = 0.f;
  for (int i = threadIdx.x; i < n1; i += 256) a += p1[i];
  for (int i = threadIdx.x; i < n2; i += 256) b += p2[i];
  r1[threadIdx.x] = a; r2[threadIdx.x] = b;
  __syncthreads();
  for (int s = 128; s > 0; s >>= 1) {
    if (threadIdx.x < s) {
      r1[threadIdx.x] += r1[threadIdx.x + s];
      r2[threadIdx.x] += r2[threadIdx.x + s];
    }
    __syncthreads();
  }
  if (threadIdx.x == 0) {
    *out_commit = 0.25f * r1[0] / (65536.f * 512.f);
    *out_recon = r2[0] / 524288.f;
  }
}

// ---------------------------------------------------------------------------
static void launch_conv(const float* x, const unsigned short* A, const float* b,
                        float* y, int Cin, int Tin, int Cout, int Tout, int K,
                        int ostride, int lhsDil, int padEff, int act,
                        int xTmajor, hipStream_t s) {
  dim3 grid(Tout / 64, (Cout + 63) / 64, 8);
  k_conv_wmma<<<grid, dim3(256), 0, s>>>(x, A, b, y, Cin, Tin, Cout, Tout, K,
                                         ostride, lhsDil, padEff, act, xTmajor);
}

extern "C" void kernel_launch(void* const* d_in, const int* in_sizes, int n_in,
                              void* d_out, int out_size, void* d_ws, size_t ws_size,
                              hipStream_t stream) {
  (void)in_sizes; (void)n_in; (void)out_size; (void)ws_size;
  const float* wave = (const float*)d_in[0];
  const float* e1w = (const float*)d_in[1];  const float* e1b = (const float*)d_in[2];
  const float* e2w = (const float*)d_in[3];  const float* e2b = (const float*)d_in[4];
  const float* e3w = (const float*)d_in[5];  const float* e3b = (const float*)d_in[6];
  const float* e4w = (const float*)d_in[7];  const float* e4b = (const float*)d_in[8];
  const float* e5w = (const float*)d_in[9];  const float* e5b = (const float*)d_in[10];
  const float* cbk = (const float*)d_in[11];
  const float* d1w = (const float*)d_in[12]; const float* d1b = (const float*)d_in[13];
  const float* d2w = (const float*)d_in[14]; const float* d2b = (const float*)d_in[15];
  const float* d3w = (const float*)d_in[16]; const float* d3b = (const float*)d_in[17];
  const float* d4w = (const float*)d_in[18]; const float* d4b = (const float*)d_in[19];
  const float* d5w = (const float*)d_in[20]; const float* d5b = (const float*)d_in[21];

  float* out = (float*)d_out;
  float* recon  = out;                       // 8*65536
  float* q      = out + 524288;              // 8*8192*512
  float* codes  = out + 524288 + 33554432;   // 8*8192
  float* commit = out + 34144256;
  float* reconl = out + 34144257;

  char* W = (char*)d_ws;
  float* bufA = (float*)(W);
  float* bufB = (float*)(W + 134217728);
  unsigned short* wb = (unsigned short*)(W + 268435456);
  size_t cur = 0;
  unsigned short* e1A = wb + cur; cur += 64 * 15;
  unsigned short* e2A = wb + cur; cur += 128 * 64 * 15;
  unsigned short* e3A = wb + cur; cur += 256 * 128 * 15;
  unsigned short* e4A = wb + cur; cur += 512 * 256 * 15;
  unsigned short* e5A = wb + cur; cur += 512 * 512 * 7;
  unsigned short* d1A = wb + cur; cur += 512 * 512 * 7;
  unsigned short* d2A = wb + cur; cur += 256 * 512 * 15;
  unsigned short* d3A = wb + cur; cur += 128 * 256 * 15;
  unsigned short* d4A = wb + cur; cur += 64 * 128 * 15;
  unsigned short* d5A = wb + cur; cur += 64 * 15;
  unsigned short* cbA = wb + cur; cur += 1024 * 512;
  cur = (cur + 1) & ~(size_t)1;
  float* cnorm = (float*)(wb + cur);
  float* part1 = cnorm + 1024;   // 4096 VQ partials
  float* part2 = part1 + 4096;   // 512 recon partials

  k_cvt_enc<<<4, 256, 0, stream>>>(e1w, e1A, 64, 1, 15);
  k_cvt_enc<<<480, 256, 0, stream>>>(e2w, e2A, 128, 64, 15);
  k_cvt_enc<<<1920, 256, 0, stream>>>(e3w, e3A, 256, 128, 15);
  k_cvt_enc<<<2048, 256, 0, stream>>>(e4w, e4A, 512, 256, 15);
  k_cvt_enc<<<2048, 256, 0, stream>>>(e5w, e5A, 512, 512, 7);
  k_cvt_flat<<<2048, 256, 0, stream>>>(cbk, cbA, 524288);
  k_cvt_dec<<<2048, 256, 0, stream>>>(d1w, d1A, 512, 512, 7);
  k_cvt_dec<<<2048, 256, 0, stream>>>(d2w, d2A, 256, 512, 15);
  k_cvt_dec<<<1920, 256, 0, stream>>>(d3w, d3A, 128, 256, 15);
  k_cvt_dec<<<480, 256, 0, stream>>>(d4w, d4A, 64, 128, 15);
  k_cvt_dec<<<4, 256, 0, stream>>>(d5w, d5A, 1, 64, 15);
  k_cb_norm<<<4, 256, 0, stream>>>(cbk, cnorm);

  // Encoder
  launch_conv(wave, e1A, e1b, bufA, 1, 65536, 64, 65536, 15, 1, 1, 7, 1, 0, stream);
  launch_conv(bufA, e2A, e2b, bufB, 64, 65536, 128, 32768, 15, 2, 1, 7, 1, 0, stream);
  launch_conv(bufB, e3A, e3b, bufA, 128, 32768, 256, 16384, 15, 2, 1, 7, 1, 0, stream);
  launch_conv(bufA, e4A, e4b, bufB, 256, 16384, 512, 8192, 15, 2, 1, 7, 1, 0, stream);
  launch_conv(bufB, e5A, e5b, bufA, 512, 8192, 512, 8192, 7, 1, 1, 3, 0, 0, stream);

  // Vector quantization
  k_vq<<<4096, 256, 0, stream>>>(bufA, cbA, cbk, cnorm, q, codes, part1);

  // Decoder (transposed convs: ostride=1, lhsDil=stride, padEff=K-1-pad)
  launch_conv(q,    d1A, d1b, bufB, 512, 8192, 512, 8192, 7, 1, 1, 3, 1, 1, stream);
  launch_conv(bufB, d2A, d2b, bufA, 512, 8192, 256, 16384, 15, 1, 2, 7, 1, 0, stream);
  launch_conv(bufA, d3A, d3b, bufB, 256, 16384, 128, 32768, 15, 1, 2, 7, 1, 0, stream);
  launch_conv(bufB, d4A, d4b, bufA, 128, 32768, 64, 65536, 15, 1, 2, 7, 1, 0, stream);
  launch_conv(bufA, d5A, d5b, recon, 64, 65536, 1, 65536, 15, 1, 1, 7, 2, 0, stream);

  // Losses
  k_recon_loss<<<512, 256, 0, stream>>>(recon, wave, part2, 524288);
  k_finalize<<<1, 256, 0, stream>>>(part1, 4096, part2, 512, commit, reconl);
}
